// AttentionLayer_618475291270
// MI455X (gfx1250) — matrix operand order repaired and checked by reading, not compiled, not run on hardware
//
#include <hip/hip_runtime.h>
#include <hip/hip_bf16.h>

// ---------------- problem constants ----------------
#define B_    4
#define L_    2048
#define D_    512
#define H_    8
#define DH_   64
#define BH_   32          // B*H
#define KTOP_ 38
#define KPAD_ 48          // 3 tiles of 16
#define BL_   8192        // B*L
#define SCALE_ 0.125f     // 1/sqrt(64)

typedef __attribute__((ext_vector_type(16))) _Float16 v16h;
typedef __attribute__((ext_vector_type(8)))  _Float16 v8h;
typedef __attribute__((ext_vector_type(8)))  float    v8f;

union FragU { v16h v; v8h h[2]; };

// A-fragment (16x32 f16, row-major source, row stride lda elements).
// ISA 7.12.2: lane<16 holds K 0..7 / 16..23, lane>=16 holds K 8..15 / 24..31.
__device__ __forceinline__ v16h load_a(const _Float16* A, int lda, int lane) {
  int row  = lane & 15;
  int koff = (lane < 16) ? 0 : 8;
  const _Float16* p = A + (size_t)row * lda + koff;
  FragU f;
  f.h[0] = *(const v8h*)(p);
  f.h[1] = *(const v8h*)(p + 16);
  return f.v;
}

// B-fragment (32x16): B[k][n] = Bt[n][k] with Bt row-major (row stride ldb).
// lane n<16 -> Bt[n][0..7,16..23]; lane>=16 -> Bt[n-16][8..15,24..31].
__device__ __forceinline__ v16h load_b(const _Float16* Bt, int ldb, int lane) {
  int row  = lane & 15;
  int koff = (lane < 16) ? 0 : 8;
  const _Float16* p = Bt + (size_t)row * ldb + koff;
  FragU f;
  f.h[0] = *(const v8h*)(p);
  f.h[1] = *(const v8h*)(p + 16);
  return f.v;
}

__device__ __forceinline__ v8f wmma16(v16h a, v16h b, v8f c) {
  return __builtin_amdgcn_wmma_f32_16x16x32_f16(false, a, false, b, (short)0, c,
                                                false, false);
}

// ---------------- elementwise prep ----------------
__global__ void k_cvt_x(const float* __restrict__ x, _Float16* __restrict__ xh, int n) {
  int i = blockIdx.x * blockDim.x + threadIdx.x;
  if (i < n) xh[i] = (_Float16)x[i];
}

__global__ void k_transpose_w(const float* __restrict__ W, _Float16* __restrict__ WT) {
  int i = blockIdx.x * blockDim.x + threadIdx.x;   // D_*D_ threads
  int n = i / D_, k = i % D_;
  WT[(size_t)n * D_ + k] = (_Float16)W[(size_t)k * D_ + n];
}

// ---------------- Q/K/V projection GEMM (WMMA, 32x32 tile / wave) ----------------
__global__ void k_proj(const _Float16* __restrict__ xh,
                       const _Float16* __restrict__ WTq, const _Float16* __restrict__ WTk,
                       const _Float16* __restrict__ WTv,
                       const float* __restrict__ bq, const float* __restrict__ bk,
                       const float* __restrict__ bv,
                       _Float16* __restrict__ Qh, _Float16* __restrict__ Kh,
                       float* __restrict__ Vf, _Float16* __restrict__ VT) {
  int lane = threadIdx.x & 31;
  int gw = blockIdx.x * (blockDim.x >> 5) + (threadIdx.x >> 5);
  int proj = gw / (256 * 16);          // M/32 = 256, N/32 = 16
  int rem  = gw % (256 * 16);
  int mt = rem / 16, nt = rem % 16;

  const _Float16* WT   = (proj == 0) ? WTq : (proj == 1) ? WTk : WTv;
  const float*    bias = (proj == 0) ? bq  : (proj == 1) ? bk  : bv;

  const _Float16* A0 = xh + (size_t)(mt * 32) * D_;
  const _Float16* A1 = A0 + (size_t)16 * D_;
  const _Float16* B0 = WT + (size_t)(nt * 32) * D_;
  const _Float16* B1 = B0 + (size_t)16 * D_;

  v8f c00 = {}, c01 = {}, c10 = {}, c11 = {};
#pragma unroll
  for (int kk = 0; kk < D_; kk += 32) {
    v16h a0 = load_a(A0 + kk, D_, lane);
    v16h a1 = load_a(A1 + kk, D_, lane);
    v16h b0 = load_b(B0 + kk, D_, lane);
    v16h b1 = load_b(B1 + kk, D_, lane);
    c00 = wmma16(a0, b0, c00);
    c01 = wmma16(a0, b1, c01);
    c10 = wmma16(a1, b0, c10);
    c11 = wmma16(a1, b1, c11);
  }

  v8f cij[2][2] = {{c00, c01}, {c10, c11}};
#pragma unroll
  for (int i = 0; i < 2; ++i) {
#pragma unroll
    for (int j = 0; j < 2; ++j) {
      int col = nt * 32 + j * 16 + (lane & 15);
      float bc = bias[col];
      int h = col >> 6, d = col & 63;
#pragma unroll
      for (int r = 0; r < 8; ++r) {
        int m = mt * 32 + i * 16 + ((lane < 16) ? r : 8 + r);
        int b_ = m >> 11, l = m & (L_ - 1);
        int bh = b_ * H_ + h;
        float y = cij[i][j][r] + bc;
        if (proj == 2) {
          Vf[((size_t)bh * L_ + l) * DH_ + d] = y;
          VT[((size_t)bh * DH_ + d) * L_ + l] = (_Float16)y;
        } else {
          _Float16* dst = (proj == 0) ? Qh : Kh;
          dst[((size_t)bh * L_ + l) * DH_ + d] = (_Float16)y;
        }
      }
    }
  }
}

// -------- full score pass: M = rowmax - rowmean (WMMA, 32 queries / wave) --------
__global__ void k_score_stats(const _Float16* __restrict__ Qh,
                              const _Float16* __restrict__ Kh,
                              float* __restrict__ Mout) {
  int lane = threadIdx.x & 31;
  int gw = blockIdx.x * (blockDim.x >> 5) + (threadIdx.x >> 5); // BH_*64 waves
  int bh = gw >> 6;
  int qt = gw & 63;                    // 64 tiles of 32 queries

  const _Float16* Qtile = Qh + ((size_t)bh * L_ + qt * 32) * DH_;
  const _Float16* Khead = Kh + (size_t)bh * L_ * DH_;
  // resident A fragments: queries [0..15] and [16..31], K-halves 0/32
  v16h a00 = load_a(Qtile, DH_, lane);
  v16h a01 = load_a(Qtile + 32, DH_, lane);
  v16h a10 = load_a(Qtile + (size_t)16 * DH_, DH_, lane);
  v16h a11 = load_a(Qtile + (size_t)16 * DH_ + 32, DH_, lane);

  float rmax[2][8], rsum[2][8];
#pragma unroll
  for (int i = 0; i < 2; ++i)
#pragma unroll
    for (int r = 0; r < 8; ++r) { rmax[i][r] = -1e30f; rsum[i][r] = 0.f; }

  for (int kt = 0; kt < L_ / 16; ++kt) {
    const _Float16* Kt = Khead + (size_t)kt * 16 * DH_;
    __builtin_prefetch(Kt + 16 * DH_, 0, 1);     // global_prefetch_b8 next tile
    v16h b0 = load_b(Kt, DH_, lane);
    v16h b1 = load_b(Kt + 32, DH_, lane);
    v8f c0 = {}, c1 = {};
    c0 = wmma16(a00, b0, c0);
    c0 = wmma16(a01, b1, c0);
    c1 = wmma16(a10, b0, c1);
    c1 = wmma16(a11, b1, c1);
#pragma unroll
    for (int r = 0; r < 8; ++r) {
      rmax[0][r] = fmaxf(rmax[0][r], c0[r]);  rsum[0][r] += c0[r];
      rmax[1][r] = fmaxf(rmax[1][r], c1[r]);  rsum[1][r] += c1[r];
    }
  }
  // reduce across the 16 key-lanes of each half per query row
#pragma unroll
  for (int m = 8; m >= 1; m >>= 1) {
#pragma unroll
    for (int i = 0; i < 2; ++i)
#pragma unroll
      for (int r = 0; r < 8; ++r) {
        rmax[i][r] = fmaxf(rmax[i][r], __shfl_xor(rmax[i][r], m, 32));
        rsum[i][r] += __shfl_xor(rsum[i][r], m, 32);
      }
  }
  if ((lane & 15) == 0) {
#pragma unroll
    for (int i = 0; i < 2; ++i) {
      int qbase = qt * 32 + i * 16 + ((lane < 16) ? 0 : 8);
#pragma unroll
      for (int r = 0; r < 8; ++r)
        Mout[(size_t)bh * L_ + qbase + r] = rmax[i][r] - rsum[i][r] * (1.0f / L_);
    }
  }
}

// ---------------- per-head iterative top-38 ----------------
__global__ void k_topk(const float* __restrict__ M, int* __restrict__ topidx) {
  __shared__ float sval[L_];
  __shared__ float rv[256];
  __shared__ int   ri[256];
  int bh = blockIdx.x, tid = threadIdx.x;
  for (int i = tid; i < L_; i += 256) sval[i] = M[(size_t)bh * L_ + i];
  __syncthreads();
  for (int t = 0; t < KTOP_; ++t) {
    float best = -1e30f; int bi = 0;
    for (int i = tid; i < L_; i += 256) {
      float v = sval[i];
      if (v > best) { best = v; bi = i; }
    }
    rv[tid] = best; ri[tid] = bi;
    __syncthreads();
    for (int s = 128; s > 0; s >>= 1) {
      if (tid < s) {
        if (rv[tid + s] > rv[tid] ||
            (rv[tid + s] == rv[tid] && ri[tid + s] < ri[tid])) {
          rv[tid] = rv[tid + s]; ri[tid] = ri[tid + s];
        }
      }
      __syncthreads();
    }
    if (tid == 0) { topidx[bh * KTOP_ + t] = ri[0]; sval[ri[0]] = -1e30f; }
    __syncthreads();
  }
}

// ---------------- gather sparse queries (pad to 48 rows) ----------------
__global__ void k_gather_qs(const _Float16* __restrict__ Qh,
                            const int* __restrict__ topidx,
                            _Float16* __restrict__ Qs) {
  int bh = blockIdx.x;
  for (int e = threadIdx.x; e < KPAD_ * DH_; e += 256) {
    int i = e >> 6, d = e & 63;
    _Float16 v = (_Float16)0.0f;
    if (i < KTOP_) {
      int idx = topidx[bh * KTOP_ + i];
      v = Qh[((size_t)bh * L_ + idx) * DH_ + d];
    }
    Qs[((size_t)bh * KPAD_ + i) * DH_ + d] = v;
  }
}

// ---------------- sparse scores S[BH,48,L] (WMMA) ----------------
__global__ void k_sparse_scores(const _Float16* __restrict__ Qs,
                                const _Float16* __restrict__ Kh,
                                float* __restrict__ S) {
  int lane = threadIdx.x & 31;
  int gw = blockIdx.x * (blockDim.x >> 5) + (threadIdx.x >> 5); // BH_*3*128
  int bh = gw / (3 * 128);
  int rem = gw % (3 * 128);
  int qt = rem / 128, kt = rem % 128;

  const _Float16* Arow = Qs + ((size_t)bh * KPAD_ + qt * 16) * DH_;
  const _Float16* Kt   = Kh + ((size_t)bh * L_ + kt * 16) * DH_;
  v16h a0 = load_a(Arow, DH_, lane);
  v16h a1 = load_a(Arow + 32, DH_, lane);
  v16h b0 = load_b(Kt, DH_, lane);
  v16h b1 = load_b(Kt + 32, DH_, lane);
  v8f c = {};
  c = wmma16(a0, b0, c);
  c = wmma16(a1, b1, c);
  int coln = kt * 16 + (lane & 15);
#pragma unroll
  for (int r = 0; r < 8; ++r) {
    int row = qt * 16 + ((lane < 16) ? r : 8 + r);
    S[((size_t)bh * KPAD_ + row) * L_ + coln] = c[r];
  }
}

// ---------------- row softmax -> P f16 (rows >= 38 zeroed) ----------------
__global__ void k_softmax(const float* __restrict__ S, _Float16* __restrict__ P) {
  __shared__ float red[256];
  int row = blockIdx.x;                 // BH_*KPAD_
  int i = row % KPAD_;
  int tid = threadIdx.x;
  const float* Srow = S + (size_t)row * L_;
  _Float16*    Prow = P + (size_t)row * L_;
  if (i >= KTOP_) {
    for (int k = tid; k < L_; k += 256) Prow[k] = (_Float16)0.0f;
    return;
  }
  float m = -1e30f;
  for (int k = tid; k < L_; k += 256) m = fmaxf(m, Srow[k] * SCALE_);
  red[tid] = m; __syncthreads();
  for (int s = 128; s > 0; s >>= 1) {
    if (tid < s) red[tid] = fmaxf(red[tid], red[tid + s]);
    __syncthreads();
  }
  m = red[0]; __syncthreads();
  float sum = 0.f;
  for (int k = tid; k < L_; k += 256) sum += __expf(Srow[k] * SCALE_ - m);
  red[tid] = sum; __syncthreads();
  for (int s = 128; s > 0; s >>= 1) {
    if (tid < s) red[tid] += red[tid + s];
    __syncthreads();
  }
  float inv = 1.0f / red[0];
  for (int k = tid; k < L_; k += 256)
    Prow[k] = (_Float16)(__expf(Srow[k] * SCALE_ - m) * inv);
}

// ---------------- context_sparse = P @ V (WMMA, B operand from V^T) ----------------
__global__ void k_pv(const _Float16* __restrict__ P, const _Float16* __restrict__ VT,
                     float* __restrict__ ctxs) {
  int lane = threadIdx.x & 31;
  int gw = blockIdx.x * (blockDim.x >> 5) + (threadIdx.x >> 5); // BH_*3*4
  int bh = gw / 12;
  int rem = gw % 12;
  int qt = rem / 4, nt = rem % 4;

  const _Float16* Arow = P + ((size_t)bh * KPAD_ + qt * 16) * L_;
  const _Float16* Bt   = VT + ((size_t)bh * DH_ + nt * 16) * L_;
  v8f c = {};
  for (int kk = 0; kk < L_; kk += 32) {
    v16h a = load_a(Arow + kk, L_, lane);
    v16h b = load_b(Bt + kk, L_, lane);
    c = wmma16(a, b, c);
  }
  int d = nt * 16 + (lane & 15);
#pragma unroll
  for (int r = 0; r < 8; ++r) {
    int row = qt * 16 + ((lane < 16) ? r : 8 + r);
    ctxs[((size_t)bh * KPAD_ + row) * DH_ + d] = c[r];
  }
}

// ---------------- V mean over L per head ----------------
__global__ void k_vmean(const float* __restrict__ Vf, float* __restrict__ vmean) {
  __shared__ float sm[256];
  int bh = blockIdx.x, tid = threadIdx.x;
  int d = tid & 63, seg = tid >> 6;     // 4 segments
  float acc = 0.f;
  for (int l = seg; l < L_; l += 4)
    acc += Vf[((size_t)bh * L_ + l) * DH_ + d];
  sm[tid] = acc; __syncthreads();
  if (tid < 64)
    vmean[bh * DH_ + tid] =
        (sm[tid] + sm[tid + 64] + sm[tid + 128] + sm[tid + 192]) * (1.0f / L_);
}

// -------- merged context: fill with V_mean, then scatter sparse rows --------
__global__ void k_fill_ctx(const float* __restrict__ vmean, _Float16* __restrict__ ctx, int n) {
  int i = blockIdx.x * blockDim.x + threadIdx.x;
  if (i >= n) return;
  int m = i / D_, col = i % D_;
  int b = m >> 11;
  int h = col >> 6, d = col & 63;
  ctx[i] = (_Float16)vmean[(b * H_ + h) * DH_ + d];
}

__global__ void k_scatter_ctx(const float* __restrict__ ctxs, const int* __restrict__ topidx,
                              _Float16* __restrict__ ctx) {
  int bh = blockIdx.x / KTOP_;
  int i  = blockIdx.x % KTOP_;
  int b = bh / H_, h = bh % H_;
  int idx = topidx[bh * KTOP_ + i];
  int d = threadIdx.x;                  // 64 threads
  ctx[((size_t)(b * L_ + idx)) * D_ + h * DH_ + d] =
      (_Float16)ctxs[((size_t)bh * KPAD_ + i) * DH_ + d];
}

// -------- output GEMM: out = ctx @ Wo + bo (WMMA, 32x32 tile / wave) --------
__global__ void k_out_gemm(const _Float16* __restrict__ ctx, const _Float16* __restrict__ WTo,
                           const float* __restrict__ bo, float* __restrict__ out) {
  int lane = threadIdx.x & 31;
  int gw = blockIdx.x * (blockDim.x >> 5) + (threadIdx.x >> 5); // 256*16 waves
  int mt = gw / 16, nt = gw % 16;
  const _Float16* A0 = ctx + (size_t)(mt * 32) * D_;
  const _Float16* A1 = A0 + (size_t)16 * D_;
  const _Float16* B0 = WTo + (size_t)(nt * 32) * D_;
  const _Float16* B1 = B0 + (size_t)16 * D_;

  v8f c00 = {}, c01 = {}, c10 = {}, c11 = {};
#pragma unroll
  for (int kk = 0; kk < D_; kk += 32) {
    v16h a0 = load_a(A0 + kk, D_, lane);
    v16h a1 = load_a(A1 + kk, D_, lane);
    v16h b0 = load_b(B0 + kk, D_, lane);
    v16h b1 = load_b(B1 + kk, D_, lane);
    c00 = wmma16(a0, b0, c00);
    c01 = wmma16(a0, b1, c01);
    c10 = wmma16(a1, b0, c10);
    c11 = wmma16(a1, b1, c11);
  }
  v8f cij[2][2] = {{c00, c01}, {c10, c11}};
#pragma unroll
  for (int i = 0; i < 2; ++i) {
#pragma unroll
    for (int j = 0; j < 2; ++j) {
      int col = nt * 32 + j * 16 + (lane & 15);
      float bc = bo[col];
#pragma unroll
      for (int r = 0; r < 8; ++r) {
        int m = mt * 32 + i * 16 + ((lane < 16) ? r : 8 + r);
        out[(size_t)m * D_ + col] = cij[i][j][r] + bc;
      }
    }
  }
}

// ---------------- host launcher ----------------
extern "C" void kernel_launch(void* const* d_in, const int* in_sizes, int n_in,
                              void* d_out, int out_size, void* d_ws, size_t ws_size,
                              hipStream_t stream) {
  (void)in_sizes; (void)n_in; (void)out_size; (void)ws_size;
  const float* x  = (const float*)d_in[0];
  const float* Wq = (const float*)d_in[1];
  const float* bq = (const float*)d_in[2];
  const float* Wk = (const float*)d_in[3];
  const float* bk = (const float*)d_in[4];
  const float* Wv = (const float*)d_in[5];
  const float* bv = (const float*)d_in[6];
  const float* Wo = (const float*)d_in[7];
  const float* bo = (const float*)d_in[8];
  float* out = (float*)d_out;

  char* p = (char*)d_ws;
  auto carve = [&](size_t bytes) -> void* {
    void* r = (void*)p;
    p += (bytes + 255) & ~(size_t)255;
    return r;
  };
  _Float16* xh   = (_Float16*)carve((size_t)BL_ * D_ * 2);
  _Float16* WTq  = (_Float16*)carve((size_t)D_ * D_ * 2);
  _Float16* WTk  = (_Float16*)carve((size_t)D_ * D_ * 2);
  _Float16* WTv  = (_Float16*)carve((size_t)D_ * D_ * 2);
  _Float16* WTo  = (_Float16*)carve((size_t)D_ * D_ * 2);
  _Float16* Qh   = (_Float16*)carve((size_t)BH_ * L_ * DH_ * 2);
  _Float16* Kh   = (_Float16*)carve((size_t)BH_ * L_ * DH_ * 2);
  float*    Vf   = (float*)   carve((size_t)BH_ * L_ * DH_ * 4);
  _Float16* VT   = (_Float16*)carve((size_t)BH_ * DH_ * L_ * 2);
  float*    Mst  = (float*)   carve((size_t)BH_ * L_ * 4);
  int*      tidx = (int*)     carve((size_t)BH_ * KTOP_ * 4);
  _Float16* Qs   = (_Float16*)carve((size_t)BH_ * KPAD_ * DH_ * 2);
  float*    S    = (float*)   carve((size_t)BH_ * KPAD_ * L_ * 4);
  _Float16* P    = (_Float16*)carve((size_t)BH_ * KPAD_ * L_ * 2);
  float*    ctxs = (float*)   carve((size_t)BH_ * KPAD_ * DH_ * 4);
  float*    vmn  = (float*)   carve((size_t)BH_ * DH_ * 4);
  _Float16* ctx  = (_Float16*)carve((size_t)BL_ * D_ * 2);

  // prep
  k_cvt_x<<<(BL_ * D_) / 256, 256, 0, stream>>>(x, xh, BL_ * D_);
  k_transpose_w<<<(D_ * D_) / 256, 256, 0, stream>>>(Wq, WTq);
  k_transpose_w<<<(D_ * D_) / 256, 256, 0, stream>>>(Wk, WTk);
  k_transpose_w<<<(D_ * D_) / 256, 256, 0, stream>>>(Wv, WTv);
  k_transpose_w<<<(D_ * D_) / 256, 256, 0, stream>>>(Wo, WTo);

  // Q/K/V projections: 3 * 256 * 16 waves (32x32 tiles), 8 waves/block
  k_proj<<<(3 * 256 * 16) / 8, 256, 0, stream>>>(xh, WTq, WTk, WTv, bq, bk, bv,
                                                 Qh, Kh, Vf, VT);
  // full score statistics: BH_*64 waves (32 queries each)
  k_score_stats<<<(BH_ * 64) / 8, 256, 0, stream>>>(Qh, Kh, Mst);
  // top-38 per head
  k_topk<<<BH_, 256, 0, stream>>>(Mst, tidx);
  // sparse attention
  k_gather_qs<<<BH_, 256, 0, stream>>>(Qh, tidx, Qs);
  k_sparse_scores<<<(BH_ * 3 * 128) / 8, 256, 0, stream>>>(Qs, Kh, S);
  k_softmax<<<BH_ * KPAD_, 256, 0, stream>>>(S, P);
  k_pv<<<(BH_ * 3 * 4) / 8, 256, 0, stream>>>(P, VT, ctxs);
  // mean-fill + scatter
  k_vmean<<<BH_, 256, 0, stream>>>(Vf, vmn);
  k_fill_ctx<<<(BL_ * D_) / 256, 256, 0, stream>>>(vmn, ctx, BL_ * D_);
  k_scatter_ctx<<<BH_ * KTOP_, 64, 0, stream>>>(ctxs, tidx, ctx);
  // output projection
  k_out_gemm<<<(512 * 32) / 32, 256, 0, stream>>>(ctx, WTo, bo, out);
}
